// AttentiveStateMLP_5643587027362
// MI455X (gfx1250) — matrix operand order repaired and checked by reading, not compiled, not run on hardware
//
#include <hip/hip_runtime.h>
#include <hip/hip_bf16.h>

typedef __attribute__((ext_vector_type(16))) _Float16 v16h;
typedef __attribute__((ext_vector_type(8)))  _Float16 v8h;
typedef __attribute__((ext_vector_type(8)))  float    v8f;

union Frag { v16h v; v8h h[2]; };

// ---------------------------------------------------------------------------
// Pack a [K,N] f32 weight matrix (row = K/input dim) into f16 WMMA B-fragment
// layout: tiles of 32(K)x16(N); within a tile, lane L (0..31) holds 16 halves
// j=0..15 with  K = kt*32 + 16*(j>=8) + 8*(L>=16) + (j&7),  N = nt*16 + (L&15).
// Tile order: kt-major over nt (tile = kt*(N/16) + nt), 512 halves per tile.
// ---------------------------------------------------------------------------
__global__ void pack_b(const float* __restrict__ src, _Float16* __restrict__ dst,
                       int K, int N) {
    int tid = blockIdx.x * 256 + threadIdx.x;
    int total = K * N;
    if (tid >= total) return;
    int tile   = tid >> 9;          // 512 halves per 32x16 tile
    int within = tid & 511;
    int lane   = within >> 4;       // 0..31
    int j      = within & 15;
    int numNT  = N >> 4;
    int kt = tile / numNT;
    int nt = tile - kt * numNT;
    int k = kt * 32 + ((j >> 3) << 4) + ((lane >> 4) << 3) + (j & 7);
    int n = (nt << 4) + (lane & 15);
    dst[tid] = (_Float16)src[k * N + n];
}

// A-fragment loader: 16-bit A 16x32 layout. lane L: row m = rowBase+(L&15);
// halves 0..7 -> K = kBase + 8*(L>=16) + 0..7 ; halves 8..15 -> +16.
__device__ __forceinline__ v16h ldsA(const _Float16* lds, int rowBase, int stride,
                                     int kBase, int lane) {
    int m = rowBase + (lane & 15);
    int g = (lane >> 4) << 3;                 // 0 or 8
    Frag f;
    f.h[0] = *(const v8h*)(lds + m * stride + kBase + g);
    f.h[1] = *(const v8h*)(lds + m * stride + kBase + 16 + g);
    return f.v;
}

// B-fragment loader from packed global weights: 32 contiguous bytes per lane.
__device__ __forceinline__ v16h ldB(const _Float16* __restrict__ wpk,
                                    int tileBase, int lane) {
    const _Float16* p = wpk + tileBase + lane * 16;
    Frag f;
    f.h[0] = *(const v8h*)(p);
    f.h[1] = *(const v8h*)(p + 8);
    return f.v;
}

#define WMMA_F16(a, b, c) \
    __builtin_amdgcn_wmma_f32_16x16x32_f16(false, (a), false, (b), (short)0, (c), false, false)

struct Params {
    const float *x;
    const float *Wv, *bv, *Wm, *bm, *Wi, *bi, *Wb, *bb, *Wc, *bc, *Wf, *bf;
    const float *pv, *pm, *pi, *pb, *pc, *pf;
    const float *bqkv, *bo, *g, *beta, *bp;
    const _Float16 *wpk;    // packed f16 weights in workspace
    float *out;
};

// Packed-weight offsets (halves) in workspace:
// Pv:0  Pm:8192  Pi:16384  Pb:20480  Pc:24576  Pf:32768
// Wqkv:49152  Wo:98304  Wp:114688   (total 147456 halves = 288 KB)

__global__ __launch_bounds__(256)
void fused_attentive_mlp(Params p) {
    __shared__ __attribute__((aligned(16))) float    xsRaw[16 * 29];  //  1856 B
    __shared__ __attribute__((aligned(16))) _Float16 hidH [16 * 384]; // 12288 B
    __shared__ __attribute__((aligned(16))) _Float16 tokH [96 * 128]; // 24576 B
    __shared__ __attribute__((aligned(16))) _Float16 qkvH [96 * 384]; // 73728 B
    __shared__ __attribute__((aligned(16))) _Float16 ctxH [96 * 128]; // 24576 B
    __shared__ float    rowMu[96];
    __shared__ float    rowRs[96];
    __shared__ __attribute__((aligned(16))) _Float16 poolH[16 * 128]; //  4096 B

    const int tid  = threadIdx.x;
    const int lane = tid & 31;              // wave32
    const int wave = tid >> 5;              // 8 waves
    const int e0   = blockIdx.x * 16;       // 16 batch elements per workgroup

    // Warm L2 path for the hot packed weights (global_prefetch_b8).
    __builtin_prefetch(p.wpk + 49152 + (size_t)lane * 64, 0, 1);

    // ---- stage x[e0:e0+16, :] (contiguous 1856 B) via async global->LDS ----
    // 116 x b128 transfers, tracked by ASYNCcnt, no VGPR round-trip.
    if (tid < 116) {
        unsigned ldsa = (unsigned)(size_t)(&xsRaw[tid * 4]);     // low 32 bits = LDS offset
        const float* ga = p.x + (size_t)e0 * 29 + tid * 4;
        asm volatile("global_load_async_to_lds_b128 %0, %1, off"
                     :: "v"(ldsa), "v"(ga) : "memory");
    }
    asm volatile("s_wait_asynccnt 0" ::: "memory");
    __syncthreads();

    // ---- component encoders (VALU; K in {2..10}, negligible FLOPs) ----
    for (int idx = tid; idx < 16 * 384; idx += 256) {
        int e = idx / 384, j = idx - e * 384;
        const float *W, *b; int xoff, din, dout, col;
        if      (j < 64)  { W=p.Wv; b=p.bv; xoff=0;  din=3;  dout=64;  col=j;       }
        else if (j < 128) { W=p.Wm; b=p.bm; xoff=3;  din=5;  dout=64;  col=j-64;    }
        else if (j < 160) { W=p.Wi; b=p.bi; xoff=8;  din=2;  dout=32;  col=j-128;   }
        else if (j < 192) { W=p.Wb; b=p.bb; xoff=10; din=3;  dout=32;  col=j-160;   }
        else if (j < 256) { W=p.Wc; b=p.bc; xoff=13; din=6;  dout=64;  col=j-192;   }
        else              { W=p.Wf; b=p.bf; xoff=19; din=10; dout=128; col=j-256;   }
        float h = b[col];
        for (int k = 0; k < din; ++k) h += xsRaw[e * 29 + xoff + k] * W[k * dout + col];
        hidH[idx] = (_Float16)fmaxf(h, 0.f);
    }
    __syncthreads();

    // ---- projections to tokens: 6 x ([16,din] @ [din,128]) via WMMA ----
    {
        const int jb[6]   = {0, 64, 128, 160, 192, 256};
        const int din[6]  = {64, 64, 32, 32, 64, 128};
        const int base[6] = {0, 8192, 16384, 20480, 24576, 32768};
        const float* bias[6] = {p.pv, p.pm, p.pi, p.pb, p.pc, p.pf};
        for (int ti = wave; ti < 48; ti += 8) {       // 6 types x 8 N-tiles
            int tt = ti >> 3, nt = ti & 7;
            int nkt = din[tt] >> 5;
            v8f acc = {};
            for (int kt = 0; kt < nkt; ++kt) {
                v16h a = ldsA(hidH, 0, 384, jb[tt] + kt * 32, lane);
                v16h b = ldB(p.wpk, base[tt] + (kt * 8 + nt) * 512, lane);
                acc = WMMA_F16(a, b, acc);
            }
            int c = (nt << 4) + (lane & 15);
            float bc = bias[tt][c];
            int mAdd = (lane >> 4) << 3;
            #pragma unroll
            for (int i = 0; i < 8; ++i) {
                int m = i + mAdd;                      // element index
                tokH[(m * 6 + tt) * 128 + c] = (_Float16)(acc[i] + bc);
            }
        }
    }
    __syncthreads();

    // ---- QKV GEMM: [96,128] @ [128,384] (6 x 24 tiles, K=4) ----
    for (int ti = wave; ti < 144; ti += 8) {
        int mt = ti / 24, nt = ti - mt * 24;
        v8f acc = {};
        #pragma unroll
        for (int kt = 0; kt < 4; ++kt) {
            v16h a = ldsA(tokH, mt * 16, 128, kt * 32, lane);
            v16h b = ldB(p.wpk, 49152 + (kt * 24 + nt) * 512, lane);
            acc = WMMA_F16(a, b, acc);
        }
        int c = (nt << 4) + (lane & 15);
        float bc = p.bqkv[c];
        int mAdd = (lane >> 4) << 3;
        #pragma unroll
        for (int i = 0; i < 8; ++i) {
            int r = mt * 16 + i + mAdd;
            qkvH[r * 384 + c] = (_Float16)(acc[i] + bc);
        }
    }
    __syncthreads();

    // ---- attention: 384 (elem, head, query) items over 256 threads ----
    for (int idx = tid; idx < 384; idx += 256) {
        int e   = idx / 24;
        int rem = idx - e * 24;
        int hd  = rem / 6;
        int t   = rem - hd * 6;
        int rb = e * 6;
        int qo = hd * 32, ko = 128 + hd * 32, vo = 256 + hd * 32;

        float q[32];
        #pragma unroll
        for (int d = 0; d < 32; ++d) q[d] = (float)qkvH[(rb + t) * 384 + qo + d];
        float s[6]; float mx = -1e30f;
        for (int u = 0; u < 6; ++u) {
            float dot = 0.f;
            #pragma unroll
            for (int d = 0; d < 32; ++d) dot += q[d] * (float)qkvH[(rb + u) * 384 + ko + d];
            s[u] = dot * 0.1767766953f;            // 1/sqrt(32)
            mx = fmaxf(mx, s[u]);
        }
        float sum = 0.f;
        for (int u = 0; u < 6; ++u) { s[u] = __expf(s[u] - mx); sum += s[u]; }
        float inv = 1.f / sum;
        float cx[32];
        #pragma unroll
        for (int d = 0; d < 32; ++d) cx[d] = 0.f;
        for (int u = 0; u < 6; ++u) {
            float w = s[u] * inv;
            #pragma unroll
            for (int d = 0; d < 32; ++d) cx[d] += w * (float)qkvH[(rb + u) * 384 + vo + d];
        }
        #pragma unroll
        for (int d = 0; d < 32; ++d) ctxH[(rb + t) * 128 + qo + d] = (_Float16)cx[d];
    }
    __syncthreads();

    // ---- out_proj + residual: h = tok + ctx@Wo + bo, written back into tokH ----
    for (int ti = wave; ti < 48; ti += 8) {            // 6 x 8 tiles
        int mt = ti >> 3, nt = ti & 7;
        v8f acc = {};
        #pragma unroll
        for (int kt = 0; kt < 4; ++kt) {
            v16h a = ldsA(ctxH, mt * 16, 128, kt * 32, lane);
            v16h b = ldB(p.wpk, 98304 + (kt * 8 + nt) * 512, lane);
            acc = WMMA_F16(a, b, acc);
        }
        int c = (nt << 4) + (lane & 15);
        float bc = p.bo[c];
        int mAdd = (lane >> 4) << 3;
        #pragma unroll
        for (int i = 0; i < 8; ++i) {
            int r = mt * 16 + i + mAdd;
            float h = (float)tokH[r * 128 + c] + acc[i] + bc;
            tokH[r * 128 + c] = (_Float16)h;           // unique (r,c) per lane: safe
        }
    }
    __syncthreads();

    // ---- layernorm stats per token row ----
    if (tid < 96) {
        float s = 0.f;
        for (int c = 0; c < 128; ++c) s += (float)tokH[tid * 128 + c];
        float mu = s * (1.f / 128.f);
        float v = 0.f;
        for (int c = 0; c < 128; ++c) {
            float d = (float)tokH[tid * 128 + c] - mu;
            v += d * d;
        }
        rowMu[tid] = mu;
        rowRs[tid] = rsqrtf(v * (1.f / 128.f) + 1e-5f);
    }
    __syncthreads();

    // ---- normalize + mean-pool over 6 tokens -> poolH [16,128] f16 ----
    for (int idx = tid; idx < 2048; idx += 256) {
        int e = idx >> 7, c = idx & 127;
        float acc = 0.f;
        for (int t = 0; t < 6; ++t) {
            int r = e * 6 + t;
            acc += ((float)tokH[r * 128 + c] - rowMu[r]) * rowRs[r];
        }
        poolH[idx] = (_Float16)(acc * (1.f / 6.f) * p.g[c] + p.beta[c]);
    }
    __syncthreads();

    // ---- final GEMM: [16,128] @ [128,256] + bias, relu, store f32 ----
    for (int nt = wave; nt < 16; nt += 8) {            // 16 N-tiles, 1 M-tile
        v8f acc = {};
        #pragma unroll
        for (int kt = 0; kt < 4; ++kt) {
            v16h a = ldsA(poolH, 0, 128, kt * 32, lane);
            v16h b = ldB(p.wpk, 114688 + (kt * 16 + nt) * 512, lane);
            acc = WMMA_F16(a, b, acc);
        }
        int c = (nt << 4) + (lane & 15);
        float bc = p.bp[c];
        int mAdd = (lane >> 4) << 3;
        #pragma unroll
        for (int i = 0; i < 8; ++i) {
            int e = i + mAdd;
            p.out[(size_t)(e0 + e) * 256 + c] = fmaxf(acc[i] + bc, 0.f);
        }
    }
}

extern "C" void kernel_launch(void* const* d_in, const int* in_sizes, int n_in,
                              void* d_out, int out_size, void* d_ws, size_t ws_size,
                              hipStream_t stream) {
    const float* x    = (const float*)d_in[0];
    const float* Wv   = (const float*)d_in[1];  const float* bv   = (const float*)d_in[2];
    const float* Wm   = (const float*)d_in[3];  const float* bm   = (const float*)d_in[4];
    const float* Wi   = (const float*)d_in[5];  const float* bi   = (const float*)d_in[6];
    const float* Wb   = (const float*)d_in[7];  const float* bb   = (const float*)d_in[8];
    const float* Wc   = (const float*)d_in[9];  const float* bc   = (const float*)d_in[10];
    const float* Wf   = (const float*)d_in[11]; const float* bf   = (const float*)d_in[12];
    const float* Pv   = (const float*)d_in[13]; const float* pv   = (const float*)d_in[14];
    const float* Pm   = (const float*)d_in[15]; const float* pm   = (const float*)d_in[16];
    const float* Pi   = (const float*)d_in[17]; const float* pi   = (const float*)d_in[18];
    const float* Pb   = (const float*)d_in[19]; const float* pb   = (const float*)d_in[20];
    const float* Pc   = (const float*)d_in[21]; const float* pc   = (const float*)d_in[22];
    const float* Pf   = (const float*)d_in[23]; const float* pf   = (const float*)d_in[24];
    const float* Wqkv = (const float*)d_in[25]; const float* bqkv = (const float*)d_in[26];
    const float* Wo   = (const float*)d_in[27]; const float* bo   = (const float*)d_in[28];
    const float* g    = (const float*)d_in[29]; const float* beta = (const float*)d_in[30];
    const float* Wp   = (const float*)d_in[31]; const float* bp   = (const float*)d_in[32];

    _Float16* wpk = (_Float16*)d_ws;   // 147456 halves = 288 KB of workspace

    auto pack = [&](const float* src, int off, int K, int N) {
        int total = K * N;
        pack_b<<<(total + 255) / 256, 256, 0, stream>>>(src, wpk + off, K, N);
    };
    pack(Pv,   0,      64, 128);
    pack(Pm,   8192,   64, 128);
    pack(Pi,   16384,  32, 128);
    pack(Pb,   20480,  32, 128);
    pack(Pc,   24576,  64, 128);
    pack(Pf,   32768, 128, 128);
    pack(Wqkv, 49152, 128, 384);
    pack(Wo,   98304, 128, 128);
    pack(Wp,  114688, 128, 256);

    Params prm;
    prm.x = x;
    prm.Wv = Wv; prm.bv = bv; prm.Wm = Wm; prm.bm = bm; prm.Wi = Wi; prm.bi = bi;
    prm.Wb = Wb; prm.bb = bb; prm.Wc = Wc; prm.bc = bc; prm.Wf = Wf; prm.bf = bf;
    prm.pv = pv; prm.pm = pm; prm.pi = pi; prm.pb = pb; prm.pc = pc; prm.pf = pf;
    prm.bqkv = bqkv; prm.bo = bo; prm.g = g; prm.beta = beta; prm.bp = bp;
    prm.wpk = wpk;
    prm.out = (float*)d_out;

    int B = in_sizes[0] / 29;          // 65536
    int grid = B / 16;                 // 16 elements per workgroup
    fused_attentive_mlp<<<grid, 256, 0, stream>>>(prm);
}
